// OctreeResBlock_4612794876218
// MI455X (gfx1250) — compile-verified, matrix operand order checked by reading, not run on hardware
//
#include <hip/hip_runtime.h>
#include <hip/hip_bf16.h>

// OctreeResBlock on MI455X (gfx1250, wave32):
//   conv1x1a -> GN/ReLU -> 27-neighbor sparse conv -> GN/ReLU -> conv1x1b -> GN
//   -> residual add -> ReLU.
// GEMMs on v_wmma_f32_16x16x32_bf16. Weights staged into LDS in fragment order
// (2x ds_load_b128 per B fragment). The GN/ReLU'd input of the dominant
// 27-neighbor conv is materialized ONCE as packed bf16, so its gathered A
// fragments are two raw global_load_b128 with no conversion VALU.

typedef __attribute__((ext_vector_type(16))) __bf16 v16bf;
typedef __attribute__((ext_vector_type(8)))  float  v8f;

#define CIN  256
#define CB   64
#define COUT 256
#define KNB  27

__device__ __forceinline__ __bf16 f2bf(float f) { return (__bf16)f; }  // RNE fptrunc
__device__ __forceinline__ unsigned short bfbits(__bf16 b) {
  unsigned short s; __builtin_memcpy(&s, &b, 2); return s;
}
__device__ __forceinline__ unsigned pack2(float lo, float hi) {
  return (unsigned)bfbits(f2bf(lo)) | ((unsigned)bfbits(f2bf(hi)) << 16);
}

// ---- stage row-major f32 weights [32*nks, ldw] into fragment-ordered bf16 LDS.
// LDS layout: ((frag*32 + lane)*16 + e), frag = ks*nnt + nt. One chunk = 8
// elems = one 16B ds_store_b128 per thread-iteration.
__device__ __forceinline__ void stage_wfrags(__bf16* dst, const float* __restrict__ src,
                                             int nks, int nnt, int ldw, int tid) {
  const int nchunk = nks * nnt * 32 * 2;
  for (int i = tid; i < nchunk; i += 256) {
    const int eh   = (i & 1) << 3;                 // 0 or 8
    const int l    = (i >> 1) & 31;
    const int frag = i >> 6;
    const int ks = frag / nnt, nt = frag - ks * nnt;
    const int hi = l >> 4;
    const int k0 = ks * 32 + (eh ? 16 : 0) + hi * 8;
    const float* s = src + (size_t)k0 * ldw + nt * 16 + (l & 15);
    uint4 q;
    q.x = pack2(s[0],               s[(size_t)ldw]);
    q.y = pack2(s[2 * (size_t)ldw], s[3 * (size_t)ldw]);
    q.z = pack2(s[4 * (size_t)ldw], s[5 * (size_t)ldw]);
    q.w = pack2(s[6 * (size_t)ldw], s[7 * (size_t)ldw]);
    *reinterpret_cast<uint4*>(dst + ((size_t)frag * 32 + l) * 16 + eh) = q;
  }
}

__device__ __forceinline__ v16bf ld_frag(const __bf16* base, int frag, int lane) {
  return *reinterpret_cast<const v16bf*>(base + ((size_t)frag * 32 + lane) * 16);
}

// A fragment straight from packed bf16 memory: elems {0..7} at p[0..7],
// elems {8..15} at p[16..23]  (p = row + ks*32 + hi*8, in ushort units)
__device__ __forceinline__ v16bf ld_a_bf(const unsigned short* p) {
  union { uint4 q[2]; v16bf v; } u;
  u.q[0] = *reinterpret_cast<const uint4*>(p);
  u.q[1] = *reinterpret_cast<const uint4*>(p + 16);
  return u.v;
}

__device__ __forceinline__ void cvt4(v16bf& a, int base, float4 f) {
  a[base + 0] = f2bf(f.x); a[base + 1] = f2bf(f.y);
  a[base + 2] = f2bf(f.z); a[base + 3] = f2bf(f.w);
}
__device__ __forceinline__ float4 gnrelu4(float4 x, float4 s, float4 h) {
  float4 r;
  r.x = fmaxf(fmaf(x.x, s.x, h.x), 0.f);
  r.y = fmaxf(fmaf(x.y, s.y, h.y), 0.f);
  r.z = fmaxf(fmaf(x.z, s.z, h.z), 0.f);
  r.w = fmaxf(fmaf(x.w, s.w, h.w), 0.f);
  return r;
}
// p points at row + ks*32 + hi*8 (32B aligned); covers k offsets {0..7,16..23}
__device__ __forceinline__ v16bf make_a(const float* p) {
  v16bf a;
  cvt4(a, 0,  *(const float4*)(p));
  cvt4(a, 4,  *(const float4*)(p + 4));
  cvt4(a, 8,  *(const float4*)(p + 16));
  cvt4(a, 12, *(const float4*)(p + 20));
  return a;
}
__device__ __forceinline__ v16bf make_a_gn(const float* p, const float* scl,
                                           const float* shf) {
  v16bf a;
  cvt4(a, 0,  gnrelu4(*(const float4*)(p),      *(const float4*)(scl),      *(const float4*)(shf)));
  cvt4(a, 4,  gnrelu4(*(const float4*)(p + 4),  *(const float4*)(scl + 4),  *(const float4*)(shf + 4)));
  cvt4(a, 8,  gnrelu4(*(const float4*)(p + 16), *(const float4*)(scl + 16), *(const float4*)(shf + 16)));
  cvt4(a, 12, gnrelu4(*(const float4*)(p + 20), *(const float4*)(scl + 20), *(const float4*)(shf + 20)));
  return a;
}

__global__ void k_zero(float* p, int cnt) {
  int i = blockIdx.x * blockDim.x + threadIdx.x;
  if (i < cnt) p[i] = 0.0f;
}

// ---------------- conv1x1a: [N,256] @ [256,64] -> x1 raw, + channel stats ----
__global__ __launch_bounds__(256) void
k_conv1x1a(const float* __restrict__ data, const float* __restrict__ w1a,
           float* __restrict__ x1, float* __restrict__ gsum,
           float* __restrict__ gsq, int n) {
  __shared__ alignas(32) __bf16 BF[8 * 4 * 32 * 16];   // 32 KB, 32 fragments
  __shared__ float lsum[CB], lsq[CB];
  const int tid = threadIdx.x, lane = tid & 31, wv = tid >> 5;
  stage_wfrags(BF, w1a, 8, 4, CB, tid);
  if (tid < CB) { lsum[tid] = 0.f; lsq[tid] = 0.f; }
  __syncthreads();

  const int hi = lane >> 4, col = lane & 15;
  const int row0 = blockIdx.x * 128 + wv * 16;
  int arow = row0 + col; if (arow >= n) arow = n - 1;
  const float* dptr = data + (size_t)arow * CIN + hi * 8;

  v8f acc[4] = {};
#pragma unroll
  for (int ks = 0; ks < CIN / 32; ++ks) {
    const v16bf a = make_a(dptr + ks * 32);
#pragma unroll
    for (int nt = 0; nt < 4; ++nt) {
      const v16bf b = ld_frag(BF, ks * 4 + nt, lane);
      acc[nt] = __builtin_amdgcn_wmma_f32_16x16x32_bf16(
          false, a, false, b, (short)0, acc[nt], false, false);
    }
  }
#pragma unroll
  for (int nt = 0; nt < 4; ++nt) {
    const int c = nt * 16 + col;
    float s = 0.f, q = 0.f;
#pragma unroll
    for (int r = 0; r < 8; ++r) {
      const int row = row0 + r + hi * 8;
      const float v = acc[nt][r];
      if (row < n) { x1[(size_t)row * CB + c] = v; s += v; q += v * v; }
    }
    atomicAdd(&lsum[c], s); atomicAdd(&lsq[c], q);
  }
  __syncthreads();
  if (tid < CB) { atomicAdd(&gsum[tid], lsum[tid]); atomicAdd(&gsq[tid], lsq[tid]); }
}

// ------------- GN finalize: per-channel scale/shift from group stats ---------
__global__ void k_gn_fin(const float* __restrict__ gsum, const float* __restrict__ gsq,
                         const float* __restrict__ gamma, const float* __restrict__ beta,
                         float* __restrict__ scale, float* __restrict__ shift,
                         int C, int groups, int n) {
  const int c = threadIdx.x;
  if (c >= C) return;
  const int cpg = C / groups, g = c / cpg;
  float s = 0.f, q = 0.f;
  for (int j = 0; j < cpg; ++j) { s += gsum[g * cpg + j]; q += gsq[g * cpg + j]; }
  const float cnt = (float)n * (float)cpg;
  const float mu = s / cnt;
  const float var = q / cnt - mu * mu;
  const float r = rsqrtf(var + 1e-5f);
  const float a = gamma[c] * r;
  scale[c] = a; shift[c] = beta[c] - mu * a;
}

// ------- materialize x1bf = packed bf16(relu(gn(x1)))  [N, 64] ---------------
__global__ __launch_bounds__(256) void
k_gnrelu_bf16(const float* __restrict__ x, const float* __restrict__ scale,
              const float* __restrict__ shift, unsigned short* __restrict__ xb,
              int n) {
  const size_t i = (size_t)blockIdx.x * 256 + threadIdx.x;   // one per 8 channels
  if (i >= (size_t)n * (CB / 8)) return;
  const int c8 = (int)(i & (CB / 8 - 1)) * 8;
  const float* p = x + i * 8;
  float4 f0 = gnrelu4(*(const float4*)(p),
                      *(const float4*)(scale + c8), *(const float4*)(shift + c8));
  float4 f1 = gnrelu4(*(const float4*)(p + 4),
                      *(const float4*)(scale + c8 + 4), *(const float4*)(shift + c8 + 4));
  uint4 q;
  q.x = pack2(f0.x, f0.y); q.y = pack2(f0.z, f0.w);
  q.z = pack2(f1.x, f1.y); q.w = pack2(f1.z, f1.w);
  *reinterpret_cast<uint4*>(xb + i * 8) = q;
}

// --- octree 3x3x3 conv: gather 27 neighbors of bf16 x1bf, GEMM vs w3[k] ------
__global__ __launch_bounds__(256) void
k_conv3(const unsigned short* __restrict__ x1bf, const int* __restrict__ neigh,
        const float* __restrict__ w3, float* __restrict__ y2,
        float* __restrict__ gsum, float* __restrict__ gsq, int n) {
  __shared__ alignas(32) __bf16 BF[2 * 4 * 32 * 16];   // 8 KB, 8 fragments
  __shared__ float lsum[CB], lsq[CB];
  const int tid = threadIdx.x, lane = tid & 31, wv = tid >> 5;
  if (tid < CB) { lsum[tid] = 0.f; lsq[tid] = 0.f; }
  const int hi = lane >> 4, col = lane & 15;
  const int row0 = blockIdx.x * 128 + wv * 16;
  int arow = row0 + col; if (arow >= n) arow = n - 1;

  v8f acc[4] = {};
  for (int nb = 0; nb < KNB; ++nb) {
    __syncthreads();
    stage_wfrags(BF, w3 + (size_t)nb * CB * CB, 2, 4, CB, tid);
    __syncthreads();
    const int nrow = neigh[(size_t)arow * KNB + nb];
    const unsigned short* xp = x1bf + (size_t)nrow * CB + hi * 8;
#pragma unroll
    for (int ks = 0; ks < 2; ++ks) {
      const v16bf a = ld_a_bf(xp + ks * 32);        // 2x global_load_b128, no VALU
#pragma unroll
      for (int nt = 0; nt < 4; ++nt) {
        const v16bf b = ld_frag(BF, ks * 4 + nt, lane);
        acc[nt] = __builtin_amdgcn_wmma_f32_16x16x32_bf16(
            false, a, false, b, (short)0, acc[nt], false, false);
      }
    }
  }
#pragma unroll
  for (int nt = 0; nt < 4; ++nt) {
    const int c = nt * 16 + col;
    float s = 0.f, q = 0.f;
#pragma unroll
    for (int r = 0; r < 8; ++r) {
      const int row = row0 + r + hi * 8;
      const float v = acc[nt][r];
      if (row < n) { y2[(size_t)row * CB + c] = v; s += v; q += v * v; }
    }
    atomicAdd(&lsum[c], s); atomicAdd(&lsq[c], q);
  }
  __syncthreads();
  if (tid < CB) { atomicAdd(&gsum[tid], lsum[tid]); atomicAdd(&gsq[tid], lsq[tid]); }
}

// ------------- conv1x1b: gn_relu(y2)[N,64] @ [64,256] -> y3 raw --------------
__global__ __launch_bounds__(256) void
k_conv1x1b(const float* __restrict__ y2, const float* __restrict__ w1b,
           const float* __restrict__ sc, const float* __restrict__ sh,
           float* __restrict__ y3, float* __restrict__ gsum,
           float* __restrict__ gsq, int n) {
  __shared__ alignas(32) __bf16 BF[2 * 16 * 32 * 16];  // 32 KB, 32 fragments
  __shared__ alignas(32) float sscale[CB], sshift[CB];
  __shared__ float lsum[COUT], lsq[COUT];
  const int tid = threadIdx.x, lane = tid & 31, wv = tid >> 5;
  stage_wfrags(BF, w1b, 2, 16, COUT, tid);
  if (tid < CB) { sscale[tid] = sc[tid]; sshift[tid] = sh[tid]; }
  lsum[tid] = 0.f; lsq[tid] = 0.f;
  __syncthreads();

  const int hi = lane >> 4, col = lane & 15;
  const int row0 = blockIdx.x * 128 + wv * 16;
  int arow = row0 + col; if (arow >= n) arow = n - 1;
  const float* xp = y2 + (size_t)arow * CB + hi * 8;

  v16bf a[2];
#pragma unroll
  for (int ks = 0; ks < 2; ++ks)
    a[ks] = make_a_gn(xp + ks * 32, sscale + ks * 32 + hi * 8,
                      sshift + ks * 32 + hi * 8);
#pragma unroll
  for (int nt = 0; nt < COUT / 16; ++nt) {
    v8f acc = {};
#pragma unroll
    for (int ks = 0; ks < 2; ++ks) {
      const v16bf b = ld_frag(BF, ks * 16 + nt, lane);
      acc = __builtin_amdgcn_wmma_f32_16x16x32_bf16(
          false, a[ks], false, b, (short)0, acc, false, false);
    }
    const int c = nt * 16 + col;
    float s = 0.f, q = 0.f;
#pragma unroll
    for (int r = 0; r < 8; ++r) {
      const int row = row0 + r + hi * 8;
      const float v = acc[r];
      if (row < n) { y3[(size_t)row * COUT + c] = v; s += v; q += v * v; }
    }
    atomicAdd(&lsum[c], s); atomicAdd(&lsq[c], q);
  }
  __syncthreads();
  atomicAdd(&gsum[tid], lsum[tid]); atomicAdd(&gsq[tid], lsq[tid]);
}

// ------------- out = relu(gn(y3) + data) -------------------------------------
__global__ __launch_bounds__(256) void
k_resid(const float* __restrict__ y3, const float* __restrict__ scale,
        const float* __restrict__ shift, const float* __restrict__ data,
        float* __restrict__ out, int n) {
  const size_t i = (size_t)blockIdx.x * 256 + threadIdx.x;
  if (i >= (size_t)n * COUT) return;
  const int c = (int)(i & (COUT - 1));
  const float v = y3[i] * scale[c] + shift[c] + data[i];
  out[i] = v > 0.f ? v : 0.f;
}

extern "C" void kernel_launch(void* const* d_in, const int* in_sizes, int n_in,
                              void* d_out, int out_size, void* d_ws, size_t ws_size,
                              hipStream_t stream) {
  const float* data = (const float*)d_in[0];
  const int*   neigh = (const int*)d_in[1];
  const float* w1a = (const float*)d_in[2];
  const float* g1a = (const float*)d_in[3];
  const float* b1a = (const float*)d_in[4];
  const float* w3  = (const float*)d_in[5];
  const float* g3  = (const float*)d_in[6];
  const float* b3  = (const float*)d_in[7];
  const float* w1b = (const float*)d_in[8];
  const float* g1b = (const float*)d_in[9];
  const float* b1b = (const float*)d_in[10];
  float* out = (float*)d_out;

  const int n = in_sizes[0] / CIN;          // N = 150000

  // workspace (floats): x1[N*64] | y2[N*64] | y3[N*256] | stats/affine | x1bf
  float* ws  = (float*)d_ws;
  float* x1  = ws;
  float* y2  = x1 + (size_t)n * CB;
  float* y3  = y2 + (size_t)n * CB;
  float* st  = y3 + (size_t)n * COUT;
  float* s1 = st;         float* q1 = s1 + CB;
  float* s2 = q1 + CB;    float* q2 = s2 + CB;
  float* s3 = q2 + CB;    float* q3 = s3 + COUT;
  float* sc1 = q3 + COUT; float* sh1 = sc1 + CB;
  float* sc2 = sh1 + CB;  float* sh2 = sc2 + CB;
  float* sc3 = sh2 + CB;  float* sh3 = sc3 + COUT;
  unsigned short* x1bf = (unsigned short*)(sh3 + COUT);   // N*64 bf16

  const int nstats = 8 * CB + 4 * COUT;     // sums + affine
  k_zero<<<(nstats + 255) / 256, 256, 0, stream>>>(st, nstats);

  const int nb = (n + 127) / 128;           // 128 rows per 8-wave block
  k_conv1x1a<<<nb, 256, 0, stream>>>(data, w1a, x1, s1, q1, n);
  k_gn_fin<<<1, CB, 0, stream>>>(s1, q1, g1a, b1a, sc1, sh1, CB, 32, n);
  k_gnrelu_bf16<<<(int)(((size_t)n * (CB / 8) + 255) / 256), 256, 0, stream>>>(
      x1, sc1, sh1, x1bf, n);
  k_conv3<<<nb, 256, 0, stream>>>(x1bf, neigh, w3, y2, s2, q2, n);
  k_gn_fin<<<1, CB, 0, stream>>>(s2, q2, g3, b3, sc2, sh2, CB, 32, n);
  k_conv1x1b<<<nb, 256, 0, stream>>>(y2, w1b, sc2, sh2, y3, s3, q3, n);
  k_gn_fin<<<1, COUT, 0, stream>>>(s3, q3, g1b, b1b, sc3, sh3, COUT, 32, n);
  k_resid<<<(int)(((size_t)n * COUT + 255) / 256), 256, 0, stream>>>(
      y3, sc3, sh3, data, out, n);
}
